// SpatiotemporalAttention_59863254172595
// MI455X (gfx1250) — matrix operand order repaired and checked by reading, not compiled, not run on hardware
//
#include <hip/hip_runtime.h>

// ---------------------------------------------------------------------------
// Spatiotemporal attention, CDNA5 (gfx1250) bf16 WMMA implementation, v2.
// All WMMA operands stored K-contiguous so each fragment is 2x b128 loads.
//   Xt, Qt, Kt, Attnt : [token][channel]  (K of their GEMMs = channel axis)
//   Vd                : [channel][token]  (K of the PV GEMM = token axis)
// ---------------------------------------------------------------------------

typedef __attribute__((ext_vector_type(16))) __bf16         v16bf;
typedef __attribute__((ext_vector_type(16))) unsigned short v16u;
typedef __attribute__((ext_vector_type(4)))  unsigned int   v4u;
typedef __attribute__((ext_vector_type(8)))  float          v8f;

#define N_TOK    4096
#define C_DIM    512
#define QKV_ROWS 1536
#define HD       64
#define LOG2E    1.4426950408889634f

union Frag {
  v16u  u;
  v4u   q[2];   // two 16-byte halves (elements 0..7 / 8..15)
  v16bf b;
};

__device__ __forceinline__ unsigned short f32_to_bf16(float f) {
  unsigned int u = __float_as_uint(f);
  u += 0x7FFFu + ((u >> 16) & 1u);   // round-to-nearest-even
  return (unsigned short)(u >> 16);
}

__device__ __forceinline__ v8f wmma_bf16(v16u a, v16u b, v8f c) {
  Frag fa, fb;
  fa.u = a; fb.u = b;
  return __builtin_amdgcn_wmma_f32_16x16x32_bf16(
      false, fa.b, false, fb.b, (short)0, c, false, false);
}

// A-frag (16x32 bf16): element e -> K = e + 8*hb + (e>=8 ? 8 : 0).
// Two contiguous 8-element runs at K = 8*hb and K = 16 + 8*hb.
__device__ __forceinline__ v16u load_frag_a(const unsigned short* base, int hb) {
  Frag f;
  f.q[0] = *(const v4u*)(base + 8 * hb);
  f.q[1] = *(const v4u*)(base + 16 + 8 * hb);
  return f.u;
}

// B-frag (32x16 bf16): element e -> K = e + 8*hb + (e>=8 ? 8 : 0).
// Two contiguous 8-element runs at K = 8*hb and K = 16 + 8*hb.
__device__ __forceinline__ v16u load_frag_b(const unsigned short* base, int hb) {
  Frag f;
  f.q[0] = *(const v4u*)(base + 8 * hb);
  f.q[1] = *(const v4u*)(base + 16 + 8 * hb);
  return f.u;
}

// ------------------------- stage 0: fp32 -> bf16 ---------------------------
// Xt is written TRANSPOSED: Xt[token][channel].
__global__ void stattn_cvt_kernel(const float* __restrict__ x,
                                  const float* __restrict__ wqkv,
                                  const float* __restrict__ wout,
                                  unsigned short* __restrict__ Xt,
                                  unsigned short* __restrict__ Wqkvb,
                                  unsigned short* __restrict__ Woutb) {
  int tid = blockIdx.x * blockDim.x + threadIdx.x;
  int stride = gridDim.x * blockDim.x;
  for (int i = tid; i < C_DIM * N_TOK; i += stride) {
    int c = i >> 12, n = i & (N_TOK - 1);
    Xt[(size_t)n * C_DIM + c] = f32_to_bf16(x[i]);
  }
  for (int i = tid; i < QKV_ROWS * C_DIM; i += stride) Wqkvb[i] = f32_to_bf16(wqkv[i]);
  for (int i = tid; i < C_DIM * C_DIM; i += stride)    Woutb[i] = f32_to_bf16(wout[i]);
}

// ----------------- stage 1: qkv = Wqkv @ X -------------------------------
// One wave per 16x64 output strip (A-frag reused over 4 WMMAs).
// Rows 0..511 -> Qt[token][ch], 512..1023 -> Kt[token][ch], 1024.. -> Vd[ch][token].
__global__ __launch_bounds__(128) void stattn_gemm_qkv_kernel(
    const unsigned short* __restrict__ Wb,   // [1536,512] bf16 row-major
    const unsigned short* __restrict__ Xt,   // [4096,512] bf16 token-major
    unsigned short* __restrict__ Qt,         // [4096,512]
    unsigned short* __restrict__ Kt,         // [4096,512]
    unsigned short* __restrict__ Vd) {       // [512,4096]
  const int lane = threadIdx.x & 31;
  const int wave = threadIdx.x >> 5;
  const int id   = blockIdx.x * 4 + wave;    // 96 m-tiles * 64 n-strips
  const int m0   = (id >> 6) * 16;
  const int n0   = (id & 63) * 64;
  const int m    = lane & 15;
  const int hb   = lane >> 4;

  v8f acc[4] = {v8f{}, v8f{}, v8f{}, v8f{}};
  for (int c0 = 0; c0 < C_DIM; c0 += 32) {
    v16u a = load_frag_a(Wb + (size_t)(m0 + m) * C_DIM + c0, hb);
#pragma unroll
    for (int s = 0; s < 4; ++s) {
      v16u b = load_frag_b(Xt + (size_t)(n0 + s * 16 + m) * C_DIM + c0, hb);
      acc[s] = wmma_bf16(a, b, acc[s]);
    }
  }
#pragma unroll
  for (int s = 0; s < 4; ++s) {
    int n = n0 + s * 16 + m;
#pragma unroll
    for (int r = 0; r < 8; ++r) {
      int o = m0 + r + 8 * hb;
      unsigned short v = f32_to_bf16(acc[s][r]);
      if (m0 < C_DIM)               Qt[(size_t)n * C_DIM + o] = v;
      else if (m0 < 2 * C_DIM)      Kt[(size_t)n * C_DIM + (o - C_DIM)] = v;
      else                          Vd[(size_t)(o - 2 * C_DIM) * N_TOK + n] = v;
    }
  }
}

// ------------- stage 2: flash attention, one wave per (head, 16 queries) ---
__global__ __launch_bounds__(128) void stattn_flash_kernel(
    const unsigned short* __restrict__ Qt,     // [4096,512]
    const unsigned short* __restrict__ Kt,     // [4096,512]
    const unsigned short* __restrict__ Vd,     // [512,4096]
    unsigned short* __restrict__ Attnt) {      // [4096,512]
  const int lane = threadIdx.x & 31;
  const int wave = threadIdx.x >> 5;
  const int gw   = blockIdx.x * 4 + wave;      // 8 heads * 256 q-tiles
  const int head = gw >> 8;
  const int q0   = (gw & 255) << 4;
  const int m    = lane & 15;
  const int hb   = lane >> 4;
  const int ch0  = head * HD;

  __shared__ unsigned short pShared[4][16 * 32];   // per-wave P staging (bf16)
  unsigned short* pLds = pShared[wave];

  // Q^T tile as two A-frags covering d = 0..31 and 32..63 (K-contiguous).
  v16u aq[2];
#pragma unroll
  for (int kk = 0; kk < 2; ++kk)
    aq[kk] = load_frag_a(Qt + (size_t)(q0 + m) * C_DIM + ch0 + kk * 32, hb);

  float mrow[8], lrow[8], alpha[8];
  v8f o[4] = {v8f{}, v8f{}, v8f{}, v8f{}};
#pragma unroll
  for (int r = 0; r < 8; ++r) { mrow[r] = -1e30f; lrow[r] = 0.f; }

  for (int j0 = 0; j0 < N_TOK; j0 += 32) {
    // Prefetch next key/value block (wrapped so always in-bounds).
    int jn = (j0 + 32) & (N_TOK - 1);
    __builtin_prefetch(Kt + (size_t)(jn + m) * C_DIM + ch0, 0, 1);
    __builtin_prefetch(Vd + (size_t)(ch0 + m) * N_TOK + jn, 0, 1);

    // ---- S = (Q^T K) * hd^-0.5 : key cols [j0..j0+15], [j0+16..j0+31] ----
    v8f s0 = {}, s1 = {};
#pragma unroll
    for (int kk = 0; kk < 2; ++kk) {
      v16u b0 = load_frag_b(Kt + (size_t)(j0 + m) * C_DIM + ch0 + kk * 32, hb);
      v16u b1 = load_frag_b(Kt + (size_t)(j0 + 16 + m) * C_DIM + ch0 + kk * 32, hb);
      s0 = wmma_bf16(aq[kk], b0, s0);
      s1 = wmma_bf16(aq[kk], b1, s1);
    }

    // ---- online softmax (each row lives across one 16-lane half) ----
#pragma unroll
    for (int r = 0; r < 8; ++r) {
      float v0 = s0[r] * 0.125f;            // hd^-0.5 = 1/8
      float v1 = s1[r] * 0.125f;
      float mx = fmaxf(v0, v1);
      mx = fmaxf(mx, __shfl_xor(mx, 1, 16));
      mx = fmaxf(mx, __shfl_xor(mx, 2, 16));
      mx = fmaxf(mx, __shfl_xor(mx, 4, 16));
      mx = fmaxf(mx, __shfl_xor(mx, 8, 16));
      float mn = fmaxf(mrow[r], mx);
      alpha[r] = exp2f((mrow[r] - mn) * LOG2E);
      mrow[r]  = mn;
      float p0 = exp2f((v0 - mn) * LOG2E);
      float p1 = exp2f((v1 - mn) * LOG2E);
      float rs = p0 + p1;
      rs += __shfl_xor(rs, 1, 16);
      rs += __shfl_xor(rs, 2, 16);
      rs += __shfl_xor(rs, 4, 16);
      rs += __shfl_xor(rs, 8, 16);
      lrow[r] = lrow[r] * alpha[r] + rs;
      // Stage P tile (C-layout -> row-major LDS): row r+8*hb, cols m / m+16.
      pLds[(r + 8 * hb) * 32 + m]      = f32_to_bf16(p0);
      pLds[(r + 8 * hb) * 32 + 16 + m] = f32_to_bf16(p1);
    }

    // Reload P in A-layout: two contiguous b128 LDS reads per lane.
    Frag pf;
    pf.q[0] = *(const v4u*)(pLds + m * 32 + 8 * hb);
    pf.q[1] = *(const v4u*)(pLds + m * 32 + 16 + 8 * hb);
    v16u pa = pf.u;

    // ---- O = alpha*O + P @ V^T over 4 d-tiles of 16 ----
#pragma unroll
    for (int d = 0; d < 4; ++d) {
      v16u bv = load_frag_b(Vd + (size_t)(ch0 + d * 16 + m) * N_TOK + j0, hb);
#pragma unroll
      for (int r = 0; r < 8; ++r) o[d][r] *= alpha[r];
      o[d] = wmma_bf16(pa, bv, o[d]);
    }
  }

  // ---- epilogue: O / l -> Attnt[token][channel] ----
#pragma unroll
  for (int d = 0; d < 4; ++d)
#pragma unroll
    for (int r = 0; r < 8; ++r) {
      float val = o[d][r] / lrow[r];
      Attnt[(size_t)(q0 + r + 8 * hb) * C_DIM + ch0 + d * 16 + m] =
          f32_to_bf16(val);
    }
}

// --------- stage 3: out = Wout @ attn + bias + x (fp32 epilogue) -----------
__global__ __launch_bounds__(128) void stattn_gemm_out_kernel(
    const unsigned short* __restrict__ Wb,     // [512,512] bf16
    const unsigned short* __restrict__ Attnt,  // [4096,512] token-major
    const float* __restrict__ bias,            // [512]
    const float* __restrict__ x,               // [512,4096] fp32 residual
    float* __restrict__ out) {                 // [512,4096] fp32
  const int lane = threadIdx.x & 31;
  const int wave = threadIdx.x >> 5;
  const int id   = blockIdx.x * 4 + wave;      // 32 m-tiles * 64 n-strips
  const int m0   = (id >> 6) * 16;
  const int n0   = (id & 63) * 64;
  const int m    = lane & 15;
  const int hb   = lane >> 4;

  v8f acc[4] = {v8f{}, v8f{}, v8f{}, v8f{}};
  for (int c0 = 0; c0 < C_DIM; c0 += 32) {
    v16u a = load_frag_a(Wb + (size_t)(m0 + m) * C_DIM + c0, hb);
#pragma unroll
    for (int s = 0; s < 4; ++s) {
      v16u b = load_frag_b(Attnt + (size_t)(n0 + s * 16 + m) * C_DIM + c0, hb);
      acc[s] = wmma_bf16(a, b, acc[s]);
    }
  }
#pragma unroll
  for (int s = 0; s < 4; ++s) {
    int n = n0 + s * 16 + m;
#pragma unroll
    for (int r = 0; r < 8; ++r) {
      int oc = m0 + r + 8 * hb;
      size_t idx = (size_t)oc * N_TOK + n;
      out[idx] = acc[s][r] + bias[oc] + x[idx];
    }
  }
}

// ---------------------------------------------------------------------------
extern "C" void kernel_launch(void* const* d_in, const int* in_sizes, int n_in,
                              void* d_out, int out_size, void* d_ws, size_t ws_size,
                              hipStream_t stream) {
  const float* x    = (const float*)d_in[0];   // [1,512,4,32,32]
  const float* wqkv = (const float*)d_in[1];   // [1536,512]
  const float* wout = (const float*)d_in[2];   // [512,512]
  const float* bout = (const float*)d_in[3];   // [512]
  float* out = (float*)d_out;

  char* ws = (char*)d_ws;
  unsigned short* Xt    = (unsigned short*)(ws);                // 4 MB   [4096,512]
  unsigned short* Wqkvb = (unsigned short*)(ws + 4194304);      // 1.5 MB [1536,512]
  unsigned short* Woutb = (unsigned short*)(ws + 5767168);      // 0.5 MB [512,512]
  unsigned short* Qt    = (unsigned short*)(ws + 6291456);      // 4 MB   [4096,512]
  unsigned short* Kt    = (unsigned short*)(ws + 10485760);     // 4 MB   [4096,512]
  unsigned short* Vd    = (unsigned short*)(ws + 14680064);     // 4 MB   [512,4096]
  unsigned short* Attnt = (unsigned short*)(ws + 18874368);     // 4 MB   [4096,512]

  hipLaunchKernelGGL(stattn_cvt_kernel, dim3(2048), dim3(256), 0, stream,
                     x, wqkv, wout, Xt, Wqkvb, Woutb);
  hipLaunchKernelGGL(stattn_gemm_qkv_kernel, dim3(1536), dim3(128), 0, stream,
                     Wqkvb, Xt, Qt, Kt, Vd);
  hipLaunchKernelGGL(stattn_flash_kernel, dim3(512), dim3(128), 0, stream,
                     Qt, Kt, Vd, Attnt);
  hipLaunchKernelGGL(stattn_gemm_out_kernel, dim3(512), dim3(128), 0, stream,
                     Woutb, Attnt, bout, x, out);
}
